// PTDNetGCNAEFriendMultiplex_19387482374958
// MI455X (gfx1250) — matrix operand order repaired and checked by reading, not compiled, not run on hardware
//
#include <hip/hip_runtime.h>

// ---------------------------------------------------------------------------
// MI455X (gfx1250, wave32) implementation of PTDNet-GCN-AE multiplex forward.
// All dense math via v_wmma_f32_16x16x32_bf16.
// Edge MLP: 2 edge-tiles x 4 n-tiles register blocking (B frag shared across
// A tiles, A frag shared across N) -> ~2x L2 traffic cut vs 1x1 tiling.
// pred GEMM: 2x4 blocking + non-temporal stores (400MB write-once stream).
// Wa2 head weights staged to LDS via global_load_async_to_lds_b128 (ASYNCcnt).
// ---------------------------------------------------------------------------

typedef __attribute__((ext_vector_type(16))) __bf16 v16bf;
typedef __attribute__((ext_vector_type(8)))  __bf16 v8bf;
typedef __attribute__((ext_vector_type(8)))  float  v8f;

union AB16 { v16bf v; v8bf h8[2]; };

// A fragment (16x32 bf16, row-major source): lane (L&15) = row M.
// Lanes 0-15 hold K = kt*32 + [0..16); lanes 16-31 get +16 offset.
__device__ __forceinline__ v16bf load_a_row(const __bf16* rowp, int kt, int lane) {
  const int kb = kt * 32 + ((lane & 16) ? 16 : 0);
  return *(const v16bf*)(rowp + kb);
}

// B fragment from pre-packed weights: 32 B contiguous per lane.
__device__ __forceinline__ v16bf load_b_packed(const __bf16* p, int kt, int nt,
                                               int ntn, int lane) {
  return *(const v16bf*)(p + ((((kt * ntn + nt) * 32) + lane) << 4));
}

__device__ __forceinline__ v8f wmma_bf16(v16bf a, v16bf b, v8f c) {
  return __builtin_amdgcn_wmma_f32_16x16x32_bf16(false, a, false, b,
                                                 (short)0, c, false, false);
}

// Async DMA: 16B global -> LDS, tracked by ASYNCcnt (per-lane addresses).
__device__ __forceinline__ void async_g2l_b128(unsigned lds_off, const void* gptr) {
  asm volatile("global_load_async_to_lds_b128 %0, %1, off"
               :: "v"(lds_off), "v"(gptr) : "memory");
}
__device__ __forceinline__ void wait_asynccnt0() {
  asm volatile("s_wait_asynccnt 0x0" ::: "memory");
}

// ------------------------- weight packing ----------------------------------
// dst[((kt*ntn+nt)*32 + lane)*16 + h] = W[kt*32 + (lane&16?16:0) + h][nt*16 + (lane&15)]
__global__ void ptd_pack_b(const float* __restrict__ W, __bf16* __restrict__ dst,
                           int K, int Nn) {
  const int tid = blockIdx.x * blockDim.x + threadIdx.x;
  const int total = (K >> 5) * (Nn >> 4) * 512;
  if (tid >= total) return;
  const int h = tid & 15;
  const int lane = (tid >> 4) & 31;
  const int tile = tid >> 9;
  const int ntn = Nn >> 4;
  const int nt = tile % ntn;
  const int kt = tile / ntn;
  const int krow = kt * 32 + ((lane & 16) ? 16 : 0) + h;
  const int col = nt * 16 + (lane & 15);
  dst[tid] = (__bf16)W[(size_t)krow * Nn + col];
}

// Wa2 [64,1] padded to a 64x16 B (only column 0 valid).
__global__ void ptd_pack_wa2(const float* __restrict__ Wa2, __bf16* __restrict__ dst) {
  const int tid = blockIdx.x * blockDim.x + threadIdx.x;
  if (tid >= 1024) return;
  const int h = tid & 15;
  const int lane = (tid >> 4) & 31;
  const int kt = tid >> 9;
  const int krow = kt * 32 + ((lane & 16) ? 16 : 0) + h;
  const int col = lane & 15;
  dst[tid] = (col == 0) ? (__bf16)Wa2[krow] : (__bf16)0.f;
}

// ------------------------- elementwise helpers -----------------------------
__global__ void ptd_fill_f32(float* p, float v, int n) {
  int i = blockIdx.x * blockDim.x + threadIdx.x;
  if (i < n) p[i] = v;
}
__global__ void ptd_cvt_bf16(const float* __restrict__ s, __bf16* __restrict__ d, int n) {
  int i = blockIdx.x * blockDim.x + threadIdx.x;
  if (i < n) d[i] = (__bf16)s[i];
}
__global__ void ptd_relu_cvt_bf16(const float* __restrict__ s, __bf16* __restrict__ d, int n) {
  int i = blockIdx.x * blockDim.x + threadIdx.x;
  if (i < n) { float v = s[i]; d[i] = (__bf16)(v > 0.f ? v : 0.f); }
}
__global__ void ptd_emb_add(float* __restrict__ acc, const float* __restrict__ x, int n) {
  int i = blockIdx.x * blockDim.x + threadIdx.x;
  if (i < n) acc[i] += x[i];
}
__global__ void ptd_finalize(const float* __restrict__ acc, float* __restrict__ emb,
                             __bf16* __restrict__ ebb, int n) {
  int i = blockIdx.x * blockDim.x + threadIdx.x;
  if (i < n) { float v = acc[i] * 0.5f; emb[i] = v; ebb[i] = (__bf16)v; }
}

// ------------------------- degree / dinv -----------------------------------
__global__ void ptd_rowsum_edges(const int* __restrict__ rows, const float* __restrict__ mask,
                                 float* __restrict__ rowsum, int E) {
  int e = blockIdx.x * blockDim.x + threadIdx.x;
  if (e < E) atomicAdd(&rowsum[rows[e]], mask[e]);
}
__global__ void ptd_dinv(const float* __restrict__ rowsum, float* __restrict__ dinv, int n) {
  int i = blockIdx.x * blockDim.x + threadIdx.x;
  if (i < n) {
    float d = rsqrtf(rowsum[i]);          // rowsum >= 1 (self loop)
    dinv[i] = fminf(fmaxf(d, 0.f), 10.f);
  }
}

// ------------------------- aggregation -------------------------------------
__global__ void ptd_agg_self(const float* __restrict__ xw, const float* __restrict__ dinv,
                             float* __restrict__ out, int N, int dout) {
  int t = blockIdx.x * blockDim.x + threadIdx.x;
  if (t >= N * dout) return;
  int i = t / dout;
  float d = dinv[i];
  out[t] = d * d * xw[t];
}
__global__ void ptd_agg_edges(const int* __restrict__ rows, const int* __restrict__ cols,
                              const float* __restrict__ mask, const float* __restrict__ dinv,
                              const float* __restrict__ xw, float* __restrict__ out, int dout) {
  const int e = blockIdx.x;
  const int r = rows[e], c = cols[e];
  const float v = mask[e] * dinv[r] * dinv[c];
  const int j = threadIdx.x;
  atomicAdd(&out[(size_t)r * dout + j], v * xw[(size_t)c * dout + j]);
}

// ------------------------- dense GEMM: C[N,Nn] = A[N,K] @ Bpacked ----------
__global__ __launch_bounds__(256)
void ptd_gemm_bf16(const __bf16* __restrict__ A, const __bf16* __restrict__ Bp,
                   float* __restrict__ C, int Nrows, int K, int Nn) {
  const int lane = threadIdx.x & 31;
  const int wtile = blockIdx.x * (blockDim.x >> 5) + (threadIdx.x >> 5);
  const int ntn = Nn >> 4;
  const int mt = wtile / ntn;
  const int nt = wtile % ntn;
  if (mt * 16 >= Nrows) return;
  const __bf16* arow = A + (size_t)(mt * 16 + (lane & 15)) * K;
  v8f c = {};
  const int ktm = K >> 5;
  for (int kt = 0; kt < ktm; ++kt) {
    v16bf a = load_a_row(arow, kt, lane);
    v16bf b = load_b_packed(Bp, kt, nt, ntn, lane);
    c = wmma_bf16(a, b, c);
  }
  const int col = nt * 16 + (lane & 15);
  const int r0 = (lane & 16) ? 8 : 0;
#pragma unroll
  for (int rr = 0; rr < 8; ++rr)
    C[(size_t)(mt * 16 + r0 + rr) * Nn + col] = c[rr];
}

// ------------------------- edge attention MLP ------------------------------
// blockDim = 64 (2 waves). Each wave owns a PAIR of 16-edge tiles (32 edges):
//  phase 1: f1/f2 with 2(M-tiles) x 4(N-tiles) register blocking, relu'd bf16
//           results to per-wave LDS [2][16x256]
//  phase 2: h = relu(fcat @ Wa1 + ba1) per tile      -> LDS [16x64]
//  phase 3: la = h @ Wa2pad (Wa2 async-staged in LDS); mask from C column 0
__global__ __launch_bounds__(64)
void ptd_edge_mlp(const __bf16* __restrict__ xb, int K,
                  const int* __restrict__ rows, const int* __restrict__ cols,
                  const __bf16* __restrict__ Wnbp, const float* __restrict__ bnb,
                  const __bf16* __restrict__ Wselfp, const float* __restrict__ bself,
                  const __bf16* __restrict__ Wa1p, const float* __restrict__ ba1,
                  const __bf16* __restrict__ Wa2p, const float* __restrict__ ba2,
                  float* __restrict__ maskOut, int npairs) {
  __shared__ __bf16 lds_f[2][2][16 * 256];   // [wave][tile] concat(f1,f2)
  __shared__ __bf16 lds_h[2][16 * 64];       // [wave] h
  __shared__ __bf16 lds_wa2[1024];           // Wa2 packed 64x16 (2 KB)

  const int tid = threadIdx.x;
  const int lane = tid & 31;
  const int wv = tid >> 5;

  // ---- async DMA stage of Wa2 into LDS (ASYNCcnt path) ----
  {
    const unsigned lbase = (unsigned)(size_t)(&lds_wa2[0]);
#pragma unroll
    for (int it = 0; it < 2; ++it) {
      const int off = (tid + it * 64) * 16;     // bytes, 2048 total
      async_g2l_b128(lbase + off, (const char*)Wa2p + off);
    }
    wait_asynccnt0();
  }
  __syncthreads();

  const int pair = blockIdx.x * 2 + wv;
  if (pair >= npairs) return;
  const int e0 = pair * 32;
  const int ktm = K >> 5;
  const int r0 = (lane & 16) ? 8 : 0;
  const int lcol = lane & 15;

  // ---- phase 1: f1 (rows/Wnb) then f2 (cols/Wself), 2x4 blocked ----
  for (int part = 0; part < 2; ++part) {
    const int* idx = part ? cols : rows;
    const __bf16* Wp = part ? Wselfp : Wnbp;
    const float* bias = part ? bself : bnb;
    const __bf16* rowp0 = xb + (size_t)idx[e0 + lcol] * K;
    const __bf16* rowp1 = xb + (size_t)idx[e0 + 16 + lcol] * K;
    const int colbase = part * 128;
    for (int ntb = 0; ntb < 2; ++ntb) {
      v8f acc[2][4] = {};
      for (int kt = 0; kt < ktm; ++kt) {
        v16bf a0 = load_a_row(rowp0, kt, lane);
        v16bf a1 = load_a_row(rowp1, kt, lane);
#pragma unroll
        for (int j = 0; j < 4; ++j) {
          v16bf b = load_b_packed(Wp, kt, ntb * 4 + j, 8, lane);
          acc[0][j] = wmma_bf16(a0, b, acc[0][j]);
          acc[1][j] = wmma_bf16(a1, b, acc[1][j]);
        }
      }
#pragma unroll
      for (int j = 0; j < 4; ++j) {
        const int col = (ntb * 4 + j) * 16 + lcol;
        const float bv = bias[col];
#pragma unroll
        for (int rr = 0; rr < 8; ++rr) {
          float v0 = acc[0][j][rr] + bv;
          float v1 = acc[1][j][rr] + bv;
          v0 = v0 > 0.f ? v0 : 0.f;
          v1 = v1 > 0.f ? v1 : 0.f;
          lds_f[wv][0][(r0 + rr) * 256 + colbase + col] = (__bf16)v0;
          lds_f[wv][1][(r0 + rr) * 256 + colbase + col] = (__bf16)v1;
        }
      }
    }
  }
  __builtin_amdgcn_wave_barrier();

  // ---- phases 2 & 3 per tile of the pair ----
  for (int t = 0; t < 2; ++t) {
    const __bf16* fbuf = lds_f[wv][t];
    __bf16* hbuf = lds_h[wv];

    // h = relu(fcat @ Wa1 + ba1), K=256, Nout=64
    for (int nt = 0; nt < 4; ++nt) {
      v8f c = {};
      for (int kt = 0; kt < 8; ++kt) {
        v16bf a = load_a_row(fbuf + lcol * 256, kt, lane);
        v16bf b = load_b_packed(Wa1p, kt, nt, 4, lane);
        c = wmma_bf16(a, b, c);
      }
      const int col = nt * 16 + lcol;
      const float bv = ba1[col];
#pragma unroll
      for (int rr = 0; rr < 8; ++rr) {
        float v = c[rr] + bv;
        v = v > 0.f ? v : 0.f;
        hbuf[(r0 + rr) * 64 + col] = (__bf16)v;
      }
    }
    __builtin_amdgcn_wave_barrier();

    // la = h @ Wa2pad (Wa2 resident in LDS; only C column 0 valid)
    v8f c = {};
#pragma unroll
    for (int kt = 0; kt < 2; ++kt) {
      v16bf a = load_a_row(hbuf + lcol * 64, kt, lane);
      v16bf b = load_b_packed((const __bf16*)lds_wa2, kt, 0, 1, lane);
      c = wmma_bf16(a, b, c);
    }
    if (lcol == 0) {                    // lanes 0 (rows 0-7) and 16 (rows 8-15)
      const float b2 = ba2[0];
      const int base = e0 + t * 16 + r0;
#pragma unroll
      for (int rr = 0; rr < 8; ++rr) {
        float la = c[rr] + b2;
        float s = 1.f / (1.f + __expf(-la));
        float mk = s * 1.2f - 0.1f;     // ZETA - GAMMA = 1.2, GAMMA = -0.1
        mk = fminf(fmaxf(mk, 0.f), 1.f);
        maskOut[base + rr] = mk;
      }
    }
    __builtin_amdgcn_wave_barrier();
  }
}

// ------------------------- pred = emb @ emb^T ------------------------------
// 2(M) x 4(N) register blocking; B column n is row n of emb (contiguous).
// Loads clamp to row N-1; stores uniformly guarded; non-temporal stores keep
// the 400MB pred stream out of L2.
__global__ __launch_bounds__(256)
void ptd_pred(const __bf16* __restrict__ Eb, float* __restrict__ P, int Nrows) {
  const int lane = threadIdx.x & 31;
  const int wunit = blockIdx.x * (blockDim.x >> 5) + (threadIdx.x >> 5);
  const int tpr = Nrows >> 4;              // 625 tiles per row/col
  const int mtP = (tpr + 1) >> 1;          // 313 row pair-blocks
  const int ntP = (tpr + 3) >> 2;          // 157 col quad-blocks
  if (wunit >= mtP * ntP) return;
  const int mb = wunit / ntP;
  const int nb = wunit % ntP;
  const int lcol = lane & 15;
  const int nclamp = Nrows - 1;

  int m0 = mb * 32 + lcol;
  int m1 = m0 + 16;
  m0 = m0 < nclamp ? m0 : nclamp;
  m1 = m1 < nclamp ? m1 : nclamp;
  const __bf16* ar0 = Eb + (size_t)m0 * 128;
  const __bf16* ar1 = Eb + (size_t)m1 * 128;
  const int kbB = (lane & 16) ? 16 : 0;

  v8f acc[2][4] = {};
#pragma unroll
  for (int kt = 0; kt < 4; ++kt) {
    v16bf a0 = load_a_row(ar0, kt, lane);
    v16bf a1 = load_a_row(ar1, kt, lane);
#pragma unroll
    for (int j = 0; j < 4; ++j) {
      int n = nb * 64 + j * 16 + lcol;
      n = n < nclamp ? n : nclamp;
      v16bf b = *(const v16bf*)(Eb + (size_t)n * 128 + kt * 32 + kbB);
      acc[0][j] = wmma_bf16(a0, b, acc[0][j]);
      acc[1][j] = wmma_bf16(a1, b, acc[1][j]);
    }
  }

  const int r0 = (lane & 16) ? 8 : 0;
#pragma unroll
  for (int t = 0; t < 2; ++t) {
    const int mrow = mb * 32 + t * 16;
    if (mrow >= Nrows) continue;                  // uniform across wave
#pragma unroll
    for (int j = 0; j < 4; ++j) {
      const int n0 = nb * 64 + j * 16;
      if (n0 >= Nrows) continue;                  // uniform across wave
      const int col = n0 + lcol;
#pragma unroll
      for (int rr = 0; rr < 8; ++rr)
        __builtin_nontemporal_store(acc[t][j][rr],
            &P[(size_t)(mrow + r0 + rr) * Nrows + col]);
    }
  }
}

// ---------------------------------------------------------------------------
static inline int cdiv(long long a, long long b) { return (int)((a + b - 1) / b); }

extern "C" void kernel_launch(void* const* d_in, const int* in_sizes, int n_in,
                              void* d_out, int out_size, void* d_ws, size_t ws_size,
                              hipStream_t stream) {
  (void)in_sizes; (void)n_in; (void)out_size; (void)ws_size;
  const int N = 10000, E = 320000, M = 2;
  const int DIN = 512, HID = 256, DOUT = 128, H1 = 128, H2 = 64;
  const int ETILES = E / 16;
  const int EPAIRS = ETILES / 2;

  const float* feats = (const float*)d_in[0];
  const int* rows = (const int*)d_in[1];
  const int* cols = (const int*)d_in[2];
  const float* Wnb0 = (const float*)d_in[3];  const float* bnb0 = (const float*)d_in[4];
  const float* Wnb1 = (const float*)d_in[5];  const float* bnb1 = (const float*)d_in[6];
  const float* Wself0 = (const float*)d_in[7]; const float* bself0 = (const float*)d_in[8];
  const float* Wself1 = (const float*)d_in[9]; const float* bself1 = (const float*)d_in[10];
  const float* Wa01 = (const float*)d_in[11]; const float* ba01 = (const float*)d_in[12];
  const float* Wa02 = (const float*)d_in[13]; const float* ba02 = (const float*)d_in[14];
  const float* Wa11 = (const float*)d_in[15]; const float* ba11 = (const float*)d_in[16];
  const float* Wa12 = (const float*)d_in[17]; const float* ba12 = (const float*)d_in[18];
  const float* Wg0 = (const float*)d_in[19];  const float* Wg1 = (const float*)d_in[20];

  // ---- workspace carve ----
  char* ws = (char*)d_ws;
  auto carve = [&](size_t bytes) -> char* {
    char* p = ws;
    ws += (bytes + 255) & ~(size_t)255;
    return p;
  };
  __bf16* xb0   = (__bf16*)carve((size_t)N * DIN * 2);   // feats bf16
  __bf16* xb1   = (__bf16*)carve((size_t)N * HID * 2);   // layer-1 input bf16
  float*  xw    = (float*) carve((size_t)N * HID * 4);   // x@Wg (max N x 256)
  float*  outb  = (float*) carve((size_t)N * HID * 4);   // aggregation output
  float*  maskw = (float*) carve((size_t)E * 4);
  float*  rowsum= (float*) carve((size_t)N * 4);
  float*  dinv  = (float*) carve((size_t)N * 4);
  float*  embs  = (float*) carve((size_t)N * DOUT * 4);
  __bf16* ebb   = (__bf16*)carve((size_t)N * DOUT * 2);
  __bf16* pWnb  = (__bf16*)carve((size_t)DIN * H1 * 2);
  __bf16* pWself= (__bf16*)carve((size_t)DIN * H1 * 2);
  __bf16* pWa1  = (__bf16*)carve((size_t)(2 * H1) * H2 * 2);
  __bf16* pWa2  = (__bf16*)carve((size_t)64 * 16 * 2);
  __bf16* pWg   = (__bf16*)carve((size_t)DIN * HID * 2);

  ptd_fill_f32<<<cdiv(N * DOUT, 256), 256, 0, stream>>>(embs, 0.f, N * DOUT);
  ptd_cvt_bf16<<<cdiv((long long)N * DIN, 256), 256, 0, stream>>>(feats, xb0, N * DIN);

  for (int m = 0; m < M; ++m) {
    for (int layer = 0; layer < 2; ++layer) {
      const int K    = layer ? HID : DIN;
      const int dout = layer ? DOUT : HID;
      const __bf16* xin = layer ? xb1 : xb0;
      const float* Wnb   = layer ? (Wnb1   + (size_t)m * HID * H1) : (Wnb0   + (size_t)m * DIN * H1);
      const float* bnb   = layer ? (bnb1   + m * H1) : (bnb0   + m * H1);
      const float* Wself = layer ? (Wself1 + (size_t)m * HID * H1) : (Wself0 + (size_t)m * DIN * H1);
      const float* bself = layer ? (bself1 + m * H1) : (bself0 + m * H1);
      const float* Wa1   = layer ? (Wa11 + (size_t)m * 2 * H1 * H2) : (Wa01 + (size_t)m * 2 * H1 * H2);
      const float* ba1   = layer ? (ba11 + m * H2) : (ba01 + m * H2);
      const float* Wa2   = layer ? (Wa12 + (size_t)m * H2) : (Wa02 + (size_t)m * H2);
      const float* ba2   = layer ? (ba12 + m) : (ba02 + m);
      const float* Wg    = layer ? (Wg1 + (size_t)m * HID * DOUT) : (Wg0 + (size_t)m * DIN * HID);

      // pack weights (bf16, B-fragment layout)
      ptd_pack_b<<<cdiv((K / 32) * (H1 / 16) * 512, 256), 256, 0, stream>>>(Wnb, pWnb, K, H1);
      ptd_pack_b<<<cdiv((K / 32) * (H1 / 16) * 512, 256), 256, 0, stream>>>(Wself, pWself, K, H1);
      ptd_pack_b<<<cdiv((256 / 32) * (H2 / 16) * 512, 256), 256, 0, stream>>>(Wa1, pWa1, 2 * H1, H2);
      ptd_pack_wa2<<<4, 256, 0, stream>>>(Wa2, pWa2);
      ptd_pack_b<<<cdiv((K / 32) * (dout / 16) * 512, 256), 256, 0, stream>>>(Wg, pWg, K, dout);

      // per-edge attention MLP -> mask
      ptd_edge_mlp<<<cdiv(EPAIRS, 2), 64, 0, stream>>>(
          xin, K, rows + (size_t)m * E, cols + (size_t)m * E,
          pWnb, bnb, pWself, bself, pWa1, ba1, pWa2, ba2, maskw, EPAIRS);

      // degree -> dinv
      ptd_fill_f32<<<cdiv(N, 256), 256, 0, stream>>>(rowsum, 1.f, N);
      ptd_rowsum_edges<<<cdiv(E, 256), 256, 0, stream>>>(rows + (size_t)m * E, maskw, rowsum, E);
      ptd_dinv<<<cdiv(N, 256), 256, 0, stream>>>(rowsum, dinv, N);

      // xw = x @ Wg
      ptd_gemm_bf16<<<cdiv((N / 16) * (dout / 16), 8), 256, 0, stream>>>(xin, pWg, xw, N, K, dout);

      // out = D^-1/2 (A+I) D^-1/2 xw
      ptd_agg_self<<<cdiv((long long)N * dout, 256), 256, 0, stream>>>(xw, dinv, outb, N, dout);
      ptd_agg_edges<<<E, dout, 0, stream>>>(rows + (size_t)m * E, cols + (size_t)m * E,
                                            maskw, dinv, xw, outb, dout);

      if (layer == 0) {
        ptd_relu_cvt_bf16<<<cdiv((long long)N * HID, 256), 256, 0, stream>>>(outb, xb1, N * HID);
      } else {
        ptd_emb_add<<<cdiv(N * DOUT, 256), 256, 0, stream>>>(embs, outb, N * DOUT);
      }
    }
  }

  float* emb_out = (float*)d_out;
  float* pred = emb_out + (size_t)N * DOUT;
  ptd_finalize<<<cdiv(N * DOUT, 256), 256, 0, stream>>>(embs, emb_out, ebb, N * DOUT);
  const int predUnits = 313 * 157;   // ceil(625/2) * ceil(625/4)
  ptd_pred<<<cdiv(predUnits, 8), 256, 0, stream>>>(ebb, pred, N);
}
